// SelfAttention_15341623181967
// MI455X (gfx1250) — matrix operand order repaired and checked by reading, not compiled, not run on hardware
//
#include <hip/hip_runtime.h>
#include <hip/hip_bf16.h>
#include <math.h>

// ---------------------------------------------------------------------------
// MI455X (gfx1250) self-attention, wave32 + v_wmma_f32_16x16x32_bf16.
//   N=4, S=2048, EMBED=1024, HEADS=16, HEAD_DIM=64
// ~89 GFLOP vs ~235 MB of mandatory traffic: matrix pipes (bf16 WMMA, f32
// accum) + LDS tile sharing across the workgroup + async loads to LDS.
// ---------------------------------------------------------------------------

#define NB    4
#define SEQ   2048
#define NHEAD 16
#define HD    64
#define EMB   1024

typedef __attribute__((ext_vector_type(16))) __bf16 v16bf;
typedef __attribute__((ext_vector_type(8)))  float  v8f;
typedef __attribute__((ext_vector_type(4)))  int    int4v;

// gfx1250 async global->LDS loads (ASYNCcnt), guarded so either toolchain
// still compiles; fallback is a plain vector copy.
#if defined(__has_builtin)
#  if __has_builtin(__builtin_amdgcn_global_load_async_to_lds_b128) && \
      __has_builtin(__builtin_amdgcn_s_wait_asynccnt)
#    define USE_ASYNC_LDS 1
#  endif
#endif

// builtin signature (from hipcc diagnostic): param0 = int4 in addrspace(1)
// (global / __device__), param1 = int4 in addrspace(3) (LDS), then two imms.
typedef __attribute__((address_space(1))) int4v glb_int4;
typedef __attribute__((address_space(3))) int4v lds_int4;

static __device__ __forceinline__ v8f wmma_bf16(v16bf a, v16bf b, v8f c) {
  // D = A(16x32 bf16) x B(32x16 bf16) + C(16x16 f32)
  return __builtin_amdgcn_wmma_f32_16x16x32_bf16(false, a, false, b,
                                                 (short)0, c, false, false);
}

// A-matrix (16x32, 16-bit) element->K mapping (contiguous, matches B fills):
// lane half 0 holds K 0-15, half 1 holds K 16-31.
static __device__ __forceinline__ int kfragA(int e, int half) {
  return half * 16 + e;
}

// ---------------------------------------------------------------------------
// Kernel: fp32 -> bf16 conversion (for Wo)
// ---------------------------------------------------------------------------
__global__ void cvt_bf16_kernel(const float* __restrict__ src,
                                __bf16* __restrict__ dst, int nelem) {
  int i = blockIdx.x * blockDim.x + threadIdx.x;
  if (i < nelem) dst[i] = (__bf16)src[i];
}

// ---------------------------------------------------------------------------
// Kernel: per-head projection  Y = X @ W^T
//   X: (N*S*H, 64) f32 rows (contiguous d), W: (64,64) f32 shared across heads
//   Y: bf16 laid out (N, H, S, D) for contiguous per-head attention reads.
// One wave = one token (16 consecutive rows == that token's 16 heads).
// ---------------------------------------------------------------------------
__global__ void proj_kernel(const float* __restrict__ X,
                            const float* __restrict__ W,
                            __bf16* __restrict__ Y) {
  __shared__ __bf16 wlds[64 * 64];
  for (int i = threadIdx.x; i < 64 * 64; i += blockDim.x)
    wlds[i] = (__bf16)W[i];
  __syncthreads();

  const int lane = threadIdx.x & 31;
  const int wave = threadIdx.x >> 5;
  const int l16 = lane & 15, half = lane >> 4;
  const int tile = blockIdx.x * 8 + wave;      // token index (n*S + s)
  const long rowbase = (long)tile * 16;        // first row of (N*S*H, 64) X

  v16bf a[2];
  {
    const float* xrow = X + (rowbase + l16) * 64;
#pragma unroll
    for (int kt = 0; kt < 2; ++kt)
#pragma unroll
      for (int e = 0; e < 16; ++e)
        a[kt][e] = (__bf16)xrow[kt * 32 + kfragA(e, half)];
  }

  const int n = tile / SEQ, s = tile % SEQ;
#pragma unroll
  for (int ct = 0; ct < 4; ++ct) {
    v8f c = {};
#pragma unroll
    for (int kt = 0; kt < 2; ++kt) {
      v16bf b;                                 // B[k][e] = W[e][k]
#pragma unroll
      for (int e = 0; e < 16; ++e)
        b[e] = wlds[(ct * 16 + l16) * 64 + kt * 32 + half * 16 + e];
      c = wmma_bf16(a[kt], b, c);
    }
#pragma unroll
    for (int r = 0; r < 8; ++r) {
      int h = r + half * 8;                    // row within tile == head id
      long o = (((long)(n * NHEAD + h) * SEQ + s) * HD) + ct * 16 + l16;
      Y[o] = (__bf16)c[r];
    }
  }
}

// ---------------------------------------------------------------------------
// Kernel: flash attention.
// One block (8 waves) owns (n, h, 128 q-rows); each wave one 16-row q tile.
// Per 32-key iteration the block cooperatively stages into LDS:
//   - K tile (32x64, row-major)  via async global->LDS b128 loads
//   - V tile TRANSPOSED (64x32)  so P@V B-fragments are contiguous ds_b128
// Double-buffered, one barrier per iteration.
// Mask applied BEFORE 1/sqrt(EMBED) scaling (matches reference).
// ---------------------------------------------------------------------------
__global__ void attn_kernel(const __bf16* __restrict__ Qp,
                            const __bf16* __restrict__ Kp,
                            const __bf16* __restrict__ Vp,
                            const int* __restrict__ mask,
                            __bf16* __restrict__ AO) {
  __shared__ __bf16 ktile[2][32 * 64];         // (key, d) row-major
  __shared__ __bf16 vtile[2][64 * 32];         // (d, key) transposed
  __shared__ __bf16 pbuf[8][16 * 32];          // per-wave P staging

  const int t    = threadIdx.x;
  const int lane = t & 31;
  const int wave = t >> 5;
  const int l16 = lane & 15, half = lane >> 4;

  const int blk  = blockIdx.x;                 // 0..1023
  const int qblk = blk & 15;                   // 16 q blocks of 128 rows
  const int h    = (blk >> 4) & 15;
  const int n    = blk >> 8;
  const int qt   = qblk * 8 + wave;            // this wave's 16-row q tile

  const long headoff = (long)(n * NHEAD + h) * SEQ * HD;
  const __bf16* Kbase = Kp + headoff;
  const __bf16* Vbase = Vp + headoff;
  __bf16* pl = pbuf[wave];

  // Q A-fragments (16 q-rows x 64 d), register-resident for the whole loop
  v16bf qa[2];
  {
    const __bf16* qrow = Qp + headoff + (long)(qt * 16 + l16) * HD;
#pragma unroll
    for (int kt = 0; kt < 2; ++kt)
#pragma unroll
      for (int e = 0; e < 16; ++e)
        qa[kt][e] = qrow[kt * 32 + kfragA(e, half)];
  }

  v8f acc[4] = {{}, {}, {}, {}};               // O: 16 q x 64 d (f32)
  float mrow[8], lrow[8];
#pragma unroll
  for (int r = 0; r < 8; ++r) { mrow[r] = -INFINITY; lrow[r] = 0.f; }

  const int* mbase = mask + (long)n * SEQ * SEQ;
  const float scale = 0.03125f;                // 1/sqrt(1024)

  for (int kb = 0; kb < SEQ / 32; ++kb) {
    const int buf = kb & 1;
    // ---- stage K tile (row-major): 4 KB, 16 B per thread -----------------
    {
      const __bf16* src = Kbase + (size_t)kb * 2048 + t * 8;
      __bf16* dst = &ktile[buf][t * 8];
#if USE_ASYNC_LDS
      __builtin_amdgcn_global_load_async_to_lds_b128(
          (glb_int4*)src, (lds_int4*)dst, 0, 0);
#else
      *(uint4*)dst = *(const uint4*)src;
#endif
    }
    // ---- stage V tile transposed: thread reads 8 contiguous d, scatters --
    {
      const int k  = t >> 3;                   // key row 0..31
      const int d0 = (t & 7) * 8;              // d chunk
      const uint4 vv = *(const uint4*)(Vbase + (size_t)kb * 2048 + k * 64 + d0);
      const __bf16* vb = (const __bf16*)&vv;
#pragma unroll
      for (int j = 0; j < 8; ++j)
        vtile[buf][(d0 + j) * 32 + k] = vb[j];
    }
    // prefetch next key block into L2 (global_prefetch_b8)
    if (kb + 1 < SEQ / 32) {
      __builtin_prefetch(Kbase + (size_t)(kb + 1) * 2048 + t * 8, 0, 0);
      __builtin_prefetch(Vbase + (size_t)(kb + 1) * 2048 + t * 8, 0, 0);
    }
#if USE_ASYNC_LDS
    __builtin_amdgcn_s_wait_asynccnt(0);
#endif
    __syncthreads();

    // ---- energy: two 16-key sub-tiles, mask then scale -------------------
    float ener[2][8];
#pragma unroll
    for (int sub = 0; sub < 2; ++sub) {
      const int k0 = kb * 32 + sub * 16;
      v8f c = {};
#pragma unroll
      for (int kt = 0; kt < 2; ++kt) {
        v16bf b;                               // B[d][key] = K'[key][d]
#pragma unroll
        for (int e = 0; e < 16; ++e)
          b[e] = ktile[buf][(sub * 16 + l16) * 64 + kt * 32 + half * 16 + e];
        c = wmma_bf16(qa[kt], b, c);
      }
#pragma unroll
      for (int r = 0; r < 8; ++r) {
        int q  = qt * 16 + r + half * 8;
        int kk = k0 + l16;
        int mv = mbase[(long)q * SEQ + kk];
        float e = (mv == 0) ? -1e20f : c[r];
        ener[sub][r] = e * scale;
      }
    }

    // ---- online softmax (rows live across 16-lane halves; xor<=8) --------
#pragma unroll
    for (int r = 0; r < 8; ++r) {
      float vmax = fmaxf(ener[0][r], ener[1][r]);
#pragma unroll
      for (int m = 8; m >= 1; m >>= 1)
        vmax = fmaxf(vmax, __shfl_xor(vmax, m, 32));
      float mnew  = fmaxf(mrow[r], vmax);
      float alpha = __expf(mrow[r] - mnew);
      float p0 = __expf(ener[0][r] - mnew);
      float p1 = __expf(ener[1][r] - mnew);
      float ps = p0 + p1;
#pragma unroll
      for (int m = 8; m >= 1; m >>= 1)
        ps += __shfl_xor(ps, m, 32);
      lrow[r] = lrow[r] * alpha + ps;
      mrow[r] = mnew;
#pragma unroll
      for (int ct = 0; ct < 4; ++ct)
        acc[ct][r] *= alpha;
      pl[(r + half * 8) * 32 + l16]      = (__bf16)p0;
      pl[(r + half * 8) * 32 + 16 + l16] = (__bf16)p1;
    }

    // ---- P A-fragment reload (compiler inserts s_wait_dscnt) -------------
    v16bf pa;
#pragma unroll
    for (int e = 0; e < 16; ++e)
      pa[e] = pl[l16 * 32 + kfragA(e, half)];

    // ---- O += P(16x32) @ V'(32x64): contiguous reads from transposed tile
#pragma unroll
    for (int ct = 0; ct < 4; ++ct) {
      v16bf bv;                                // B[k][d] = Vt[d][k]
#pragma unroll
      for (int e = 0; e < 16; ++e)
        bv[e] = vtile[buf][(ct * 16 + l16) * 32 + half * 16 + e];
      acc[ct] = wmma_bf16(pa, bv, acc[ct]);
    }
    // double-buffered: the single barrier above is the only sync needed
  }

  // ---- normalize and store as (N, S, H, D) == (N, S, EMBED) bf16 ---------
#pragma unroll
  for (int ct = 0; ct < 4; ++ct)
#pragma unroll
    for (int r = 0; r < 8; ++r) {
      int q = qt * 16 + r + half * 8;
      float o = acc[ct][r] / lrow[r];
      long off = (((long)(n * SEQ + q) * NHEAD) + h) * HD + ct * 16 + l16;
      AO[off] = (__bf16)o;
    }
}

// ---------------------------------------------------------------------------
// Kernel: output projection  out = AO(8192x1024) @ Wo^T + bo   (f32 out)
// One wave = 64 rows x 64 cols (4 row-fragments reuse each B fragment).
// 512 WMMA per wave; 4x less Wo traffic than a 16-row tile.
// ---------------------------------------------------------------------------
__global__ void outproj_kernel(const __bf16* __restrict__ AO,
                               const __bf16* __restrict__ WoB,
                               const float* __restrict__ bo,
                               float* __restrict__ out) {
  const int lane = threadIdx.x & 31;
  const int wave = threadIdx.x >> 5;
  const int l16 = lane & 15, half = lane >> 4;
  const int task = blockIdx.x * 8 + wave;      // 2048 tasks
  const int rg = task >> 4;                    // 128 row groups of 64
  const int cg = task & 15;                    // 16 col groups of 64

  v8f acc[4][4] = {{{}, {}, {}, {}}, {{}, {}, {}, {}},
                   {{}, {}, {}, {}}, {{}, {}, {}, {}}};

  for (int kb = 0; kb < EMB / 32; ++kb) {
    v16bf a[4];
#pragma unroll
    for (int rt = 0; rt < 4; ++rt) {
      const __bf16* arow = AO + (long)(rg * 64 + rt * 16 + l16) * EMB;
#pragma unroll
      for (int e = 0; e < 16; ++e)
        a[rt][e] = arow[kb * 32 + kfragA(e, half)];
    }
#pragma unroll
    for (int c = 0; c < 4; ++c) {
      v16bf b;                                 // B[k][e] = Wo[e][k]
      const __bf16* wrow = WoB + (long)(cg * 64 + c * 16 + l16) * EMB;
#pragma unroll
      for (int e = 0; e < 16; ++e)
        b[e] = wrow[kb * 32 + half * 16 + e];
#pragma unroll
      for (int rt = 0; rt < 4; ++rt)
        acc[rt][c] = wmma_bf16(a[rt], b, acc[rt][c]);
    }
  }

#pragma unroll
  for (int c = 0; c < 4; ++c) {
    int col = cg * 64 + c * 16 + l16;
    float bias = bo[col];
#pragma unroll
    for (int rt = 0; rt < 4; ++rt)
#pragma unroll
      for (int r = 0; r < 8; ++r) {
        int row = rg * 64 + rt * 16 + r + half * 8;
        out[(long)row * EMB + col] = acc[rt][c][r] + bias;
      }
  }
}

// ---------------------------------------------------------------------------
extern "C" void kernel_launch(void* const* d_in, const int* in_sizes, int n_in,
                              void* d_out, int out_size, void* d_ws, size_t ws_size,
                              hipStream_t stream) {
  const float* values  = (const float*)d_in[0];
  const float* queries = (const float*)d_in[1];
  const float* keys    = (const float*)d_in[2];
  const int*   mask    = (const int*)d_in[3];
  const float* Wv = (const float*)d_in[4];
  const float* Wk = (const float*)d_in[5];
  const float* Wq = (const float*)d_in[6];
  const float* Wo = (const float*)d_in[7];
  const float* bo = (const float*)d_in[8];

  // workspace carve-up (bf16): Q' K' V' (N,H,S,D) + attn-out (N,S,E) + Wo
  const size_t SZ = (size_t)NB * NHEAD * SEQ * HD;   // 8388608 elements
  __bf16* qp  = (__bf16*)d_ws;
  __bf16* kp  = qp + SZ;
  __bf16* vp  = kp + SZ;
  __bf16* ao  = vp + SZ;
  __bf16* wob = ao + SZ;

  cvt_bf16_kernel<<<(EMB * EMB + 255) / 256, 256, 0, stream>>>(Wo, wob, EMB * EMB);

  // projections: 8192 token-tiles, 8 waves/block
  proj_kernel<<<NB * SEQ / 8, 256, 0, stream>>>(queries, Wq, qp);
  proj_kernel<<<NB * SEQ / 8, 256, 0, stream>>>(keys,    Wk, kp);
  proj_kernel<<<NB * SEQ / 8, 256, 0, stream>>>(values,  Wv, vp);

  // attention: one block per (n, h, 128-q block) = 4*16*16 = 1024 blocks
  attn_kernel<<<NB * NHEAD * (SEQ / 128), 256, 0, stream>>>(qp, kp, vp, mask, ao);

  // output projection: 128 row-groups x 16 col-groups = 2048 wave-tasks
  outproj_kernel<<<(NB * SEQ / 64) * (EMB / 64) / 8, 256, 0, stream>>>(
      ao, wob, bo, (float*)d_out);
}
